// TransformerBlock_73203422593686
// MI455X (gfx1250) — matrix operand order repaired and checked by reading, not compiled, not run on hardware
//
#include <hip/hip_runtime.h>
#include <cstdint>
#include <cstddef>

typedef _Float16 fp16;
typedef fp16  v16h __attribute__((ext_vector_type(16)));
typedef float v8f  __attribute__((ext_vector_type(8)));
typedef int   v4i  __attribute__((ext_vector_type(4)));

#define BB     64
#define SEQ    197
#define DIM    768
#define HEADS  12
#define HDIM   64
#define MLPD   3072
#define RHD    512
#define RANKD  256
#define TTOK   (BB*SEQ)      /* 12608, divisible by 64 */
#define SPAD   224           /* padded key stride for transposed V */
#define QTILES 13            /* ceil(197/16) */

#if defined(__gfx1250__) && __has_builtin(__builtin_amdgcn_global_load_async_to_lds_b128)
#define USE_ASYNC 1
#else
#define USE_ASYNC 0
#endif

union Frag { v16h h; uint4 u[2]; };

static __device__ __forceinline__ v8f wmma32(v16h a, v16h b, v8f c) {
  // D = A(16x32 f16) * B(32x16 f16) + C(16x16 f32)
  return __builtin_amdgcn_wmma_f32_16x16x32_f16(false, a, false, b, (short)0, c, false, false);
}
static __device__ __forceinline__ void zfrag(Frag& f) {
  f.u[0] = make_uint4(0u,0u,0u,0u);
  f.u[1] = make_uint4(0u,0u,0u,0u);
}

#if USE_ASYNC
typedef __attribute__((address_space(1))) v4i* gas_v4i;
typedef __attribute__((address_space(3))) v4i* las_v4i;
static __device__ __forceinline__ void async_cp16(const uint4* g, uint4* l) {
  __builtin_amdgcn_global_load_async_to_lds_b128((gas_v4i)g, (las_v4i)l, 0, 0);
}
static __device__ __forceinline__ void wait_async0() {
#if __has_builtin(__builtin_amdgcn_s_wait_asynccnt)
  __builtin_amdgcn_s_wait_asynccnt(0);
#else
  asm volatile("s_wait_asynccnt 0" ::: "memory");
#endif
}
#endif

// ---------------------------------------------------------------- f32 -> f16
__global__ __launch_bounds__(256) void cvt_f16_kernel(const float* __restrict__ s,
                                                      fp16* __restrict__ d, int n) {
  int i = blockIdx.x * 256 + threadIdx.x;
  if (i < n) d[i] = (fp16)s[i];
}

// ------------------------------------------------- LayerNorm (+raw f16 copy)
__global__ __launch_bounds__(256) void ln_cvt_kernel(const float* __restrict__ x,
    const float* __restrict__ g, const float* __restrict__ bta,
    fp16* __restrict__ raw16, fp16* __restrict__ norm16) {
  __shared__ float sh[256];
  const int t = blockIdx.x, tid = threadIdx.x;
  const float* xr = x + (size_t)t * DIM;
  float v0 = xr[tid], v1 = xr[tid + 256], v2 = xr[tid + 512];
  sh[tid] = v0 + v1 + v2; __syncthreads();
  for (int s = 128; s > 0; s >>= 1) { if (tid < s) sh[tid] += sh[tid + s]; __syncthreads(); }
  float mean = sh[0] * (1.f / DIM); __syncthreads();
  float d0 = v0 - mean, d1 = v1 - mean, d2 = v2 - mean;
  sh[tid] = d0 * d0 + d1 * d1 + d2 * d2; __syncthreads();
  for (int s = 128; s > 0; s >>= 1) { if (tid < s) sh[tid] += sh[tid + s]; __syncthreads(); }
  float rstd = rsqrtf(sh[0] * (1.f / DIM) + 1e-5f);
  fp16* nr = norm16 + (size_t)t * DIM;
  nr[tid]       = (fp16)(d0 * rstd * g[tid]       + bta[tid]);
  nr[tid + 256] = (fp16)(d1 * rstd * g[tid + 256] + bta[tid + 256]);
  nr[tid + 512] = (fp16)(d2 * rstd * g[tid + 512] + bta[tid + 512]);
  if (raw16) {
    fp16* rr = raw16 + (size_t)t * DIM;
    rr[tid] = (fp16)v0; rr[tid + 256] = (fp16)v1; rr[tid + 512] = (fp16)v2;
  }
}

// --------------------------- router tail: LN(RH) + SiLU + 4 logits + routing
__global__ __launch_bounds__(256) void router_fin_kernel(const float* __restrict__ y,
    const float* __restrict__ g, const float* __restrict__ bta,
    const float* __restrict__ w2, const float* __restrict__ b2,
    int* __restrict__ idxArr) {
  __shared__ float sh[256];
  __shared__ float lg[4];
  const int t = blockIdx.x, tid = threadIdx.x;
  const float* yr = y + (size_t)t * RHD;
  float v0 = yr[tid], v1 = yr[tid + 256];
  sh[tid] = v0 + v1; __syncthreads();
  for (int s = 128; s > 0; s >>= 1) { if (tid < s) sh[tid] += sh[tid + s]; __syncthreads(); }
  float mean = sh[0] * (1.f / RHD); __syncthreads();
  float d0 = v0 - mean, d1 = v1 - mean;
  sh[tid] = d0 * d0 + d1 * d1; __syncthreads();
  for (int s = 128; s > 0; s >>= 1) { if (tid < s) sh[tid] += sh[tid + s]; __syncthreads(); }
  float rstd = rsqrtf(sh[0] * (1.f / RHD) + 1e-5f); __syncthreads();
  float a0 = d0 * rstd * g[tid]       + bta[tid];
  float a1 = d1 * rstd * g[tid + 256] + bta[tid + 256];
  a0 = a0 / (1.f + __expf(-a0));
  a1 = a1 / (1.f + __expf(-a1));
  for (int j = 0; j < 4; ++j) {
    sh[tid] = a0 * w2[j * RHD + tid] + a1 * w2[j * RHD + tid + 256];
    __syncthreads();
    for (int s = 128; s > 0; s >>= 1) { if (tid < s) sh[tid] += sh[tid + s]; __syncthreads(); }
    if (tid == 0) lg[j] = sh[0] + b2[j];
    __syncthreads();
  }
  if (tid == 0) {
    int s = t % SEQ;
    int id;
    if (s < 1) id = 3;  // reserved tokens -> full-one key (identity)
    else id = ((lg[1] > lg[0]) ? 2 : 0) + ((lg[3] > lg[2]) ? 1 : 0);
    idxArr[t] = id;
  }
}

// ----------------------------------------------------------- WMMA tiled GEMM
// C[M,N] = A[M,K](f16) * W[N,K](f16)^T (+bias) ; templated epilogue.
// 64x128 block tile, 8 waves (each 32x32), BK=32, async double-buffered LDS.
enum { M_F16OUT = 0, M_F32OUT, M_VSCAT, M_WO, M_GELU, M_FC2, M_DOWN, M_UPACC };

struct GemmP {
  const fp16* A; const fp16* W; const float* bias;
  void* out; const float* res; const int* idx; fp16* out2;
  int N, K, expert;
};

template <int MODE>
__global__ __launch_bounds__(256) void gemm_wmma_kernel(GemmP p) {
  __shared__ uint4 ldsA[2][256];  // 64 rows  x 32 f16 (64B/row) x2 buffers
  __shared__ uint4 ldsB[2][512];  // 128 rows x 32 f16 (64B/row) x2 buffers
  const int tid = threadIdx.x;
  const int lane = tid & 31, w = tid >> 5;
  const bool hi = lane >= 16;
  const int lr = lane & 15;
  const int m0 = blockIdx.y * 64, n0 = blockIdx.x * 128;
  const int wm = (w >> 2) * 32, wn = (w & 3) * 32;
  const int K8 = p.K >> 3;
  const uint4* gA = (const uint4*)p.A;
  const uint4* gB = (const uint4*)p.W;
  // per-thread tile chunk coordinates (16B granularity, 64B per row)
  const int ra = tid >> 2,          ca = tid & 3;          // A: chunk tid
  const int rb0 = tid >> 2,         cb0 = tid & 3;         // B: chunk tid
  const int rb1 = (tid + 256) >> 2, cb1 = tid & 3;         // B: chunk tid+256
  const size_t gaOff = (size_t)(m0 + ra) * K8 + ca;
  const size_t gb0Off = (size_t)(n0 + rb0) * K8 + cb0;
  const size_t gb1Off = (size_t)(n0 + rb1) * K8 + cb1;

  v8f z = {};
  v8f acc[2][2];
  acc[0][0] = z; acc[0][1] = z; acc[1][0] = z; acc[1][1] = z;

#if USE_ASYNC
  // prefetch tile 0 into buffer 0
  async_cp16(gA + gaOff, &ldsA[0][tid]);
  async_cp16(gB + gb0Off, &ldsB[0][tid]);
  async_cp16(gB + gb1Off, &ldsB[0][tid + 256]);
#endif

  int it = 0;
  for (int k0 = 0; k0 < p.K; k0 += 32, ++it) {
#if USE_ASYNC
    const int cur = it & 1;
    wait_async0();
    __syncthreads();
    if (k0 + 32 < p.K) {
      const int nk = (k0 + 32) >> 3;
      async_cp16(gA + gaOff + nk, &ldsA[cur ^ 1][tid]);
      async_cp16(gB + gb0Off + nk, &ldsB[cur ^ 1][tid]);
      async_cp16(gB + gb1Off + nk, &ldsB[cur ^ 1][tid + 256]);
    }
#else
    const int cur = 0;
    const int kk = k0 >> 3;
    __syncthreads();
    ldsA[0][tid]       = gA[gaOff + kk];
    ldsB[0][tid]       = gB[gb0Off + kk];
    ldsB[0][tid + 256] = gB[gb1Off + kk];
    __syncthreads();
#endif
    Frag bf[2];
#pragma unroll
    for (int ns = 0; ns < 2; ++ns) {
      int col = wn + ns * 16 + lr;
      bf[ns].u[0] = ldsB[cur][col * 4 + (hi ? 1 : 0)];
      bf[ns].u[1] = ldsB[cur][col * 4 + (hi ? 3 : 2)];
    }
#pragma unroll
    for (int ms = 0; ms < 2; ++ms) {
      int row = wm + ms * 16 + lr;
      Frag af;
      af.u[0] = ldsA[cur][row * 4 + (hi ? 1 : 0)];
      af.u[1] = ldsA[cur][row * 4 + (hi ? 3 : 2)];
      acc[ms][0] = wmma32(af.h, bf[0].h, acc[ms][0]);
      acc[ms][1] = wmma32(af.h, bf[1].h, acc[ms][1]);
    }
  }

#pragma unroll
  for (int ms = 0; ms < 2; ++ms)
#pragma unroll
    for (int ns = 0; ns < 2; ++ns) {
      int gn = n0 + wn + ns * 16 + lr;
      float bv = p.bias ? p.bias[gn] : 0.f;
#pragma unroll
      for (int r = 0; r < 8; ++r) {
        int gm = m0 + wm + ms * 16 + r + (hi ? 8 : 0);
        float v = acc[ms][ns][r] + bv;
        size_t o = (size_t)gm * p.N + gn;
        if constexpr (MODE == M_F16OUT) {
          ((fp16*)p.out)[o] = (fp16)v;
        } else if constexpr (MODE == M_F32OUT) {
          ((float*)p.out)[o] = v;
        } else if constexpr (MODE == M_VSCAT) {
          int bb = gm / SEQ, ss = gm % SEQ, hh = gn >> 6, dd = gn & 63;
          ((fp16*)p.out)[((size_t)(bb * HEADS + hh) * HDIM + dd) * SPAD + ss] = (fp16)v;
        } else if constexpr (MODE == M_WO) {
          ((float*)p.out)[o] = p.res[o] + v;
        } else if constexpr (MODE == M_GELU) {
          float gg = 0.5f * v * (1.f + erff(v * 0.70710678118f));
          ((fp16*)p.out)[o] = (fp16)gg;
        } else if constexpr (MODE == M_FC2) {
          float t = p.res[o] + v;
          ((float*)p.out)[o] = t;
          p.out2[o] = (fp16)t;
        } else if constexpr (MODE == M_DOWN) {
          ((fp16*)p.out)[o] = (p.idx[gm] == p.expert) ? (fp16)v : (fp16)0.f;
        } else if constexpr (MODE == M_UPACC) {
          ((float*)p.out)[o] += v;
        }
      }
    }
}

// --------------------------------------------- flash attention (wave-level)
__global__ __launch_bounds__(256) void attn_kernel(const fp16* __restrict__ q16,
    const fp16* __restrict__ k16, const fp16* __restrict__ vT,
    fp16* __restrict__ o16) {
  __shared__ __align__(16) fp16 plds[8][512];  // 1KB per wave: 16x32 f16 P tile
  const int lane = threadIdx.x & 31, w = threadIdx.x >> 5;
  const int task = blockIdx.x * 8 + w;
  if (task >= BB * HEADS * QTILES) return;
  const int tile = task % QTILES;
  const int bh = task / QTILES;
  const int b = bh / HEADS, h = bh % HEADS;
  const int m0 = tile * 16;
  const bool hi = lane >= 16;
  const int lr = lane & 15;

  // Q fragments (A layout): rows = query tokens, K = head dim (two K=32 frags)
  Frag aq0, aq1;
  {
    int qrow = m0 + lr;
    if (qrow < SEQ) {
      const uint4* qp = (const uint4*)(q16 + ((size_t)(b * SEQ + qrow) * DIM + h * HDIM));
      aq0.u[0] = qp[hi ? 1 : 0]; aq0.u[1] = qp[hi ? 3 : 2];
      aq1.u[0] = qp[hi ? 5 : 4]; aq1.u[1] = qp[hi ? 7 : 6];
    } else { zfrag(aq0); zfrag(aq1); }
  }

  float mrow[8], lrow[8];
#pragma unroll
  for (int r = 0; r < 8; ++r) { mrow[r] = -3.0e38f; lrow[r] = 0.f; }
  v8f z = {};
  v8f co[4]; co[0] = z; co[1] = z; co[2] = z; co[3] = z;

  for (int kc = 0; kc < 7; ++kc) {
    const int kbase = kc * 32;
    v8f st[2];
#pragma unroll
    for (int sub = 0; sub < 2; ++sub) {
      int kt = kbase + sub * 16 + lr;  // this lane's key column
      Frag bk0, bk1;
      if (kt < SEQ) {
        const uint4* kp = (const uint4*)(k16 + ((size_t)(b * SEQ + kt) * DIM + h * HDIM));
        bk0.u[0] = kp[hi ? 1 : 0]; bk0.u[1] = kp[hi ? 3 : 2];  // dims 0..31
        bk1.u[0] = kp[hi ? 5 : 4]; bk1.u[1] = kp[hi ? 7 : 6];  // dims 32..63
      } else { zfrag(bk0); zfrag(bk1); }
      v8f s = z;
      s = wmma32(aq0.h, bk0.h, s);
      s = wmma32(aq1.h, bk1.h, s);
      float madd = (kt < SEQ) ? 0.f : -1e30f;
#pragma unroll
      for (int r = 0; r < 8; ++r) s[r] = s[r] * 0.125f + madd;  // /sqrt(64)
      st[sub] = s;
    }
    // online softmax update (rows r + 8*hi; reductions over the 16-lane half)
#pragma unroll
    for (int r = 0; r < 8; ++r) {
      float v0 = st[0][r], v1 = st[1][r];
      float cmax = fmaxf(v0, v1);
      cmax = fmaxf(cmax, __shfl_xor(cmax, 1));
      cmax = fmaxf(cmax, __shfl_xor(cmax, 2));
      cmax = fmaxf(cmax, __shfl_xor(cmax, 4));
      cmax = fmaxf(cmax, __shfl_xor(cmax, 8));
      float mnew = fmaxf(mrow[r], cmax);
      float p0 = __expf(v0 - mnew), p1 = __expf(v1 - mnew);
      float ps = p0 + p1;
      ps += __shfl_xor(ps, 1);
      ps += __shfl_xor(ps, 2);
      ps += __shfl_xor(ps, 4);
      ps += __shfl_xor(ps, 8);
      float sc = __expf(mrow[r] - mnew);
      lrow[r] = lrow[r] * sc + ps;
      mrow[r] = mnew;
      co[0][r] *= sc; co[1][r] *= sc; co[2][r] *= sc; co[3][r] *= sc;
      int rowI = r + (hi ? 8 : 0);
      plds[w][rowI * 32 + lr]      = (fp16)p0;
      plds[w][rowI * 32 + 16 + lr] = (fp16)p1;
    }
    // reload P in A-fragment layout (wave-private LDS region, DS ops in-order)
    Frag pa;
    {
      const uint4* pp = (const uint4*)&plds[w][lr * 32];
      pa.u[0] = pp[hi ? 1 : 0];
      pa.u[1] = pp[hi ? 3 : 2];
    }
    // O += P @ V  (V transposed so B-fragments are contiguous)
#pragma unroll
    for (int ns = 0; ns < 4; ++ns) {
      const uint4* vp = (const uint4*)(vT + ((size_t)(b * HEADS + h) * HDIM + ns * 16 + lr) * SPAD + kbase);
      Frag bv;
      bv.u[0] = vp[hi ? 1 : 0];
      bv.u[1] = vp[hi ? 3 : 2];
      co[ns] = wmma32(pa.h, bv.h, co[ns]);
    }
  }
#pragma unroll
  for (int r = 0; r < 8; ++r) {
    int tok = m0 + r + (hi ? 8 : 0);
    if (tok < SEQ) {
      float inv = 1.f / lrow[r];
      fp16* op = o16 + (size_t)(b * SEQ + tok) * DIM + h * HDIM + lr;
      op[0]  = (fp16)(co[0][r] * inv);
      op[16] = (fp16)(co[1][r] * inv);
      op[32] = (fp16)(co[2][r] * inv);
      op[48] = (fp16)(co[3][r] * inv);
    }
  }
}

// ------------------------------------------------------------------- launch
extern "C" void kernel_launch(void* const* d_in, const int* in_sizes, int n_in,
                              void* d_out, int out_size, void* d_ws, size_t ws_size,
                              hipStream_t stream) {
  const float* x     = (const float*)d_in[0];
  const float* wq_w  = (const float*)d_in[2];  const float* wq_b = (const float*)d_in[3];
  const float* wk_w  = (const float*)d_in[4];  const float* wk_b = (const float*)d_in[5];
  const float* wv_w  = (const float*)d_in[6];  const float* wv_b = (const float*)d_in[7];
  const float* wo_w  = (const float*)d_in[8];  const float* wo_b = (const float*)d_in[9];
  const float* an_g  = (const float*)d_in[10]; const float* an_b = (const float*)d_in[11];
  const float* fc1_w = (const float*)d_in[12]; const float* fc1_b = (const float*)d_in[13];
  const float* fc2_w = (const float*)d_in[14]; const float* fc2_b = (const float*)d_in[15];
  const float* fn_g  = (const float*)d_in[16]; const float* fn_b = (const float*)d_in[17];
  const float* r1_w  = (const float*)d_in[18]; const float* r1_b = (const float*)d_in[19];
  const float* rn_g  = (const float*)d_in[20]; const float* rn_b = (const float*)d_in[21];
  const float* r2_w  = (const float*)d_in[22]; const float* r2_b = (const float*)d_in[23];
  const float* eA    = (const float*)d_in[24];
  const float* eB    = (const float*)d_in[25];
  float* out = (float*)d_out;
  char* ws = (char*)d_ws;

  const size_t szTD2 = (size_t)TTOK * DIM * 2;         // f16 [T,768]
  // region0: x16|h16|q16|k16, later a1 (T*3072*2 == 4*szTD2), later down
  size_t o_x16 = 0, o_h16 = szTD2, o_q16 = 2 * szTD2, o_k16 = 3 * szTD2;
  size_t o_a1 = 0, o_down = 0;
  size_t off = 4 * szTD2;
  auto nxt = [&](size_t b) { size_t r = off; off += (b + 255) & ~(size_t)255; return r; };
  size_t o_vT  = nxt((size_t)BB * HEADS * HDIM * SPAD * 2);
  size_t o_o16 = nxt(szTD2);
  size_t o_x1  = nxt((size_t)TTOK * DIM * 4);
  size_t o_r1o = nxt((size_t)TTOK * RHD * 4);          // later x2_f16
  size_t o_x2h = o_r1o;
  size_t o_idx = nxt((size_t)TTOK * 4);
  size_t o_n2  = nxt(szTD2);
  size_t o_wq  = nxt((size_t)DIM * DIM * 2);
  size_t o_wk  = nxt((size_t)DIM * DIM * 2);
  size_t o_wv  = nxt((size_t)DIM * DIM * 2);
  size_t o_wo  = nxt((size_t)DIM * DIM * 2);
  size_t o_f1  = nxt((size_t)MLPD * DIM * 2);
  size_t o_f2  = nxt((size_t)DIM * MLPD * 2);
  size_t o_r1w = nxt((size_t)RHD * DIM * 2);
  size_t o_eA  = nxt((size_t)3 * RANKD * DIM * 2);
  size_t o_eB  = nxt((size_t)3 * DIM * RANKD * 2);
  (void)ws_size; (void)in_sizes; (void)n_in; (void)out_size;

  fp16* x16  = (fp16*)(ws + o_x16);  fp16* h16 = (fp16*)(ws + o_h16);
  fp16* q16  = (fp16*)(ws + o_q16);  fp16* k16 = (fp16*)(ws + o_k16);
  fp16* vT16 = (fp16*)(ws + o_vT);   fp16* o16 = (fp16*)(ws + o_o16);
  float* x1  = (float*)(ws + o_x1);  float* r1o = (float*)(ws + o_r1o);
  fp16* x2h  = (fp16*)(ws + o_x2h);  int* idxA = (int*)(ws + o_idx);
  fp16* n2   = (fp16*)(ws + o_n2);   fp16* a1  = (fp16*)(ws + o_a1);
  fp16* down = (fp16*)(ws + o_down);
  fp16* wq16 = (fp16*)(ws + o_wq);   fp16* wk16 = (fp16*)(ws + o_wk);
  fp16* wv16 = (fp16*)(ws + o_wv);   fp16* wo16 = (fp16*)(ws + o_wo);
  fp16* f116 = (fp16*)(ws + o_f1);   fp16* f216 = (fp16*)(ws + o_f2);
  fp16* r1w16 = (fp16*)(ws + o_r1w);
  fp16* eA16 = (fp16*)(ws + o_eA);   fp16* eB16 = (fp16*)(ws + o_eB);

  auto cvt = [&](const float* s, fp16* d, int n) {
    cvt_f16_kernel<<<(n + 255) / 256, 256, 0, stream>>>(s, d, n);
  };
  cvt(wq_w, wq16, DIM * DIM);    cvt(wk_w, wk16, DIM * DIM);
  cvt(wv_w, wv16, DIM * DIM);    cvt(wo_w, wo16, DIM * DIM);
  cvt(fc1_w, f116, MLPD * DIM);  cvt(fc2_w, f216, DIM * MLPD);
  cvt(r1_w, r1w16, RHD * DIM);
  cvt(eA, eA16, 3 * RANKD * DIM); cvt(eB, eB16, 3 * DIM * RANKD);

  // LN(x) for attention + raw x as f16 for the router
  ln_cvt_kernel<<<TTOK, 256, 0, stream>>>(x, an_g, an_b, x16, h16);

  const dim3 blk(256);
  auto grid = [&](int N) { return dim3((unsigned)(N / 128), (unsigned)(TTOK / 64)); };
  GemmP p{};

  // router GEMM: x @ r1^T + b -> f32 [T,512]
  p = {x16, r1w16, r1_b, (void*)r1o, nullptr, nullptr, nullptr, RHD, DIM, 0};
  gemm_wmma_kernel<M_F32OUT><<<grid(RHD), blk, 0, stream>>>(p);
  router_fin_kernel<<<TTOK, 256, 0, stream>>>(r1o, rn_g, rn_b, r2_w, r2_b, idxA);

  // QKV
  p = {h16, wq16, wq_b, (void*)q16, nullptr, nullptr, nullptr, DIM, DIM, 0};
  gemm_wmma_kernel<M_F16OUT><<<grid(DIM), blk, 0, stream>>>(p);
  p = {h16, wk16, wk_b, (void*)k16, nullptr, nullptr, nullptr, DIM, DIM, 0};
  gemm_wmma_kernel<M_F16OUT><<<grid(DIM), blk, 0, stream>>>(p);
  p = {h16, wv16, wv_b, (void*)vT16, nullptr, nullptr, nullptr, DIM, DIM, 0};
  gemm_wmma_kernel<M_VSCAT><<<grid(DIM), blk, 0, stream>>>(p);

  // flash attention: 64*12*13 wave tasks, 8 waves per block
  attn_kernel<<<(BB * HEADS * QTILES) / 8, 256, 0, stream>>>(q16, k16, vT16, o16);

  // x1 = x + o @ wo^T + b
  p = {o16, wo16, wo_b, (void*)x1, x, nullptr, nullptr, DIM, DIM, 0};
  gemm_wmma_kernel<M_WO><<<grid(DIM), blk, 0, stream>>>(p);

  // LN(x1) -> n2
  ln_cvt_kernel<<<TTOK, 256, 0, stream>>>(x1, fn_g, fn_b, nullptr, n2);

  // a1 = gelu(n2 @ fc1^T + b)
  p = {n2, f116, fc1_b, (void*)a1, nullptr, nullptr, nullptr, MLPD, DIM, 0};
  gemm_wmma_kernel<M_GELU><<<grid(MLPD), blk, 0, stream>>>(p);

  // x2 = x1 + a1 @ fc2^T + b  -> d_out (f32) and x2h (f16)
  p = {a1, f216, fc2_b, (void*)out, x1, nullptr, x2h, DIM, MLPD, 0};
  gemm_wmma_kernel<M_FC2><<<grid(DIM), blk, 0, stream>>>(p);

  // LoRA experts: masked down-proj then accumulate up-proj into d_out
  for (int e = 0; e < 3; ++e) {
    p = {x2h, eA16 + (size_t)e * RANKD * DIM, nullptr,
         (void*)(down + (size_t)e * TTOK * RANKD), nullptr, idxA, nullptr,
         RANKD, DIM, e};
    gemm_wmma_kernel<M_DOWN><<<grid(RANKD), blk, 0, stream>>>(p);
  }
  for (int e = 0; e < 3; ++e) {
    p = {down + (size_t)e * TTOK * RANKD, eB16 + (size_t)e * DIM * RANKD, nullptr,
         (void*)out, nullptr, nullptr, nullptr, DIM, RANKD, 0};
    gemm_wmma_kernel<M_UPACC><<<grid(DIM), blk, 0, stream>>>(p);
  }
}
